// MultiHeadQuantumAttention_17549236371758
// MI455X (gfx1250) — matrix operand order repaired and checked by reading, not compiled, not run on hardware
//
#include <hip/hip_runtime.h>
#include <hip/hip_bf16.h>
#include <stdint.h>

// ---------------- shapes ----------------
constexpr int MM = 16384;   // B*N rows
constexpr int DD = 1024;    // model dim
constexpr int HH = 8;       // heads
constexpr int HD = 128;     // hilbert dim
constexpr int SS = 64;      // memory states

// ---------------- WMMA plumbing ----------------
typedef __attribute__((ext_vector_type(16))) __bf16 bf16x16;
typedef __attribute__((ext_vector_type(8)))  float  v8f;

struct B32 { uint4 lo, hi; };

static __device__ __forceinline__ bf16x16 ld_frag(const uint16_t* p0, const uint16_t* p1) {
  B32 u; u.lo = *(const uint4*)p0; u.hi = *(const uint4*)p1;
  return __builtin_bit_cast(bf16x16, u);
}
static __device__ __forceinline__ bf16x16 ld_frag32(const uint16_t* p) {
  return ld_frag(p, p + 8);
}
static __device__ __forceinline__ uint16_t f2bf(float f) {
  uint32_t u = __builtin_bit_cast(uint32_t, f);
  uint32_t r = u + 0x7FFFu + ((u >> 16) & 1u);   // round-to-nearest-even
  return (uint16_t)(r >> 16);
}
static __device__ __forceinline__ float bf2f(uint16_t b) {
  uint32_t u = ((uint32_t)b) << 16;
  return __builtin_bit_cast(float, u);
}
#define WMMA_BF16(a, b, c) \
  __builtin_amdgcn_wmma_f32_16x16x32_bf16(false, (a), false, (b), (short)0, (c), false, false)

// ================= prep kernels =================

// fp32 h -> bf16 row-major (A operand source)
__global__ void __launch_bounds__(256, 1)
k_cvt_h(const float* __restrict__ h, uint16_t* __restrict__ out) {
  size_t i = ((size_t)blockIdx.x * blockDim.x + threadIdx.x) * 4;
  float4 v = *(const float4*)(h + i);
  ushort4 o;
  o.x = f2bf(v.x); o.y = f2bf(v.y); o.z = f2bf(v.z); o.w = f2bf(v.w);
  *(ushort4*)(out + i) = o;
}

// W_real|W_imag -> per-head B-fragments: [h][kt=32][ntAll=16 (8 re + 8 im)][lane=32][16 halves]
__global__ void __launch_bounds__(256, 1)
k_pack_W(const float* __restrict__ Wr, const float* __restrict__ Wi,
         uint16_t* __restrict__ out) {
  int t = blockIdx.x * 256 + threadIdx.x;          // 131072 threads
  int lane = t & 31, nt = (t >> 5) & 15, kt = (t >> 9) & 31, hh = t >> 14;
  int n  = (nt & 7) * 16 + (lane & 15);
  int ks = kt * 32 + (lane >> 4) * 8;
  const float* src = ((nt < 8) ? Wr : Wi) + (size_t)hh * DD * HD;
  uint16_t* o = out + (size_t)t * 16;
  #pragma unroll
  for (int j = 0; j < 16; j++) o[j] = f2bf(src[(size_t)(ks + j + (j & 8)) * HD + n]);
}

// normalize complex memory states -> fp32 [re | im] planes
__global__ void __launch_bounds__(128, 1)
k_norm_mem(const float* __restrict__ mr, const float* __restrict__ mi,
           float* __restrict__ outn) {
  int bs = blockIdx.x;            // h*SS + s
  int d = threadIdx.x;            // 0..127
  size_t base = (size_t)bs * HD;
  float r = mr[base + d], q = mi[base + d];
  __shared__ float red[128];
  red[d] = r * r + q * q;
  __syncthreads();
  for (int s = 64; s > 0; s >>= 1) { if (d < s) red[d] += red[d + s]; __syncthreads(); }
  float sc = 1.0f / fmaxf(sqrtf(red[0]), 1e-12f);
  outn[base + d] = r * sc;
  outn[(size_t)HH * SS * HD + base + d] = q * sc;
}

// memH as two stacked real B operands (K=256): part0 = [mr; mi] (-> ov_re),
// part1 = [-mi; mr] (-> ov_im).  Layout: [h][part=2][kt=8][nt=4][lane][16]
__global__ void __launch_bounds__(256, 1)
k_pack_mem(const float* __restrict__ memn, uint16_t* __restrict__ out) {
  int t = blockIdx.x * 256 + threadIdx.x;          // 16384 threads
  int lane = t & 31, nt = (t >> 5) & 3, kt = (t >> 7) & 7, part = (t >> 10) & 1, hh = t >> 11;
  int s  = nt * 16 + (lane & 15);
  int ks = kt * 32 + (lane >> 4) * 8;
  const float* re = memn + (size_t)hh * SS * HD;
  const float* im = memn + (size_t)HH * SS * HD + (size_t)hh * SS * HD;
  uint16_t* o = out + (size_t)t * 16;
  #pragma unroll
  for (int j = 0; j < 16; j++) {
    int K = ks + j + (j & 8), kk = K & 127;
    float v;
    if (part == 0) v = (K < 128) ?  re[(size_t)s * HD + kk] : im[(size_t)s * HD + kk];
    else           v = (K < 128) ? -im[(size_t)s * HD + kk] : re[(size_t)s * HD + kk];
    o[j] = f2bf(v);
  }
}

// vg[h*64+s][g] = values[h,s,:] . gate_w[h*D: , g]
__global__ void __launch_bounds__(256, 1)
k_vg(const float* __restrict__ values, const float* __restrict__ gw,
     float* __restrict__ vg) {
  int hs = blockIdx.x;            // 512 blocks
  int hh = hs >> 6;
  int t = threadIdx.x;
  float acc[8] = {0};
  const float* vrow = values + (size_t)hs * DD;
  for (int d = t; d < DD; d += 256) {
    float v = vrow[d];
    const float* g = gw + ((size_t)hh * DD + d) * HH;
    #pragma unroll
    for (int gi = 0; gi < 8; gi++) acc[gi] += v * g[gi];
  }
  __shared__ float red[256][8];
  #pragma unroll
  for (int gi = 0; gi < 8; gi++) red[t][gi] = acc[gi];
  __syncthreads();
  for (int s = 128; s > 0; s >>= 1) {
    if (t < s) {
      #pragma unroll
      for (int gi = 0; gi < 8; gi++) red[t][gi] += red[t + s][gi];
    }
    __syncthreads();
  }
  if (t < 8) vg[(size_t)hs * 8 + t] = red[0][t];
}

// values stacked over (h,s) as K=512 B operand: [kt=16][nt=64][lane][16]
__global__ void __launch_bounds__(256, 1)
k_pack_V(const float* __restrict__ values, uint16_t* __restrict__ out) {
  int t = blockIdx.x * 256 + threadIdx.x;          // 32768 threads
  int lane = t & 31, nt = (t >> 5) & 63, kt = t >> 11;
  int n  = nt * 16 + (lane & 15);
  int ks = kt * 32 + (lane >> 4) * 8;
  uint16_t* o = out + (size_t)t * 16;
  #pragma unroll
  for (int j = 0; j < 16; j++) o[j] = f2bf(values[(size_t)(ks + j + (j & 8)) * DD + n]);
}

// out_w as K=1024 B operand: [kt=32][nt=64][lane][16]
__global__ void __launch_bounds__(256, 1)
k_pack_outw(const float* __restrict__ ow, uint16_t* __restrict__ out) {
  int t = blockIdx.x * 256 + threadIdx.x;          // 65536 threads
  int lane = t & 31, nt = (t >> 5) & 63, kt = t >> 11;
  int n  = nt * 16 + (lane & 15);
  int ks = kt * 32 + (lane >> 4) * 8;
  uint16_t* o = out + (size_t)t * 16;
  #pragma unroll
  for (int j = 0; j < 16; j++) o[j] = f2bf(ow[(size_t)(ks + j + (j & 8)) * DD + n]);
}

// ================= main pipeline =================

// psi = h @ [W_re | W_im] per head, fused complex L2 normalization, bf16 output (A layout).
// block: 512 thr = 16 waves (4 row-tiles x 4 column quarters); grid (MM/64, HH).
// K loop is a ping-pong unroll-by-2 pipeline: buffer A is consumed while buffer B
// preloads, then swap -- no register copies, loads overlap the matrix pipe.
__global__ void __launch_bounds__(512, 1)
k_psi(const uint16_t* __restrict__ hbf, const uint16_t* __restrict__ wpack,
      uint16_t* __restrict__ psib) {
  int tid = threadIdx.x, lane = tid & 31, w = tid >> 5;
  int mt = w & 3, cq = w >> 2;            // row tile, column quarter (64 cols each)
  int hh = blockIdx.y;
  int rowbase = blockIdx.x * 64 + mt * 16;
  const uint16_t* wp = wpack + (size_t)hh * 32 * 16 * 32 * 16;
  const uint16_t* abase = hbf + (size_t)(rowbase + (lane & 15)) * DD + (lane >> 4) * 8;

  v8f acc[4] = {};
  bf16x16 a0, a1, b0[4], b1[4];
  a0 = ld_frag(abase, abase + 16);
  #pragma unroll
  for (int nt = 0; nt < 4; nt++)
    b0[nt] = ld_frag32(wp + ((size_t)(cq * 4 + nt) * 32 + lane) * 16);

  for (int kt = 0; kt < 32; kt += 2) {
    int k1 = kt + 1;
    a1 = ld_frag(abase + k1 * 32, abase + k1 * 32 + 16);
    #pragma unroll
    for (int nt = 0; nt < 4; nt++)
      b1[nt] = ld_frag32(wp + ((size_t)(k1 * 16 + cq * 4 + nt) * 32 + lane) * 16);
    #pragma unroll
    for (int nt = 0; nt < 4; nt++)
      acc[nt] = WMMA_BF16(a0, b0[nt], acc[nt]);
    int k2 = (kt + 2 < 32) ? kt + 2 : 31;      // clamped redundant preload on last pass
    a0 = ld_frag(abase + k2 * 32, abase + k2 * 32 + 16);
    #pragma unroll
    for (int nt = 0; nt < 4; nt++)
      b0[nt] = ld_frag32(wp + ((size_t)(k2 * 16 + cq * 4 + nt) * 32 + lane) * 16);
    #pragma unroll
    for (int nt = 0; nt < 4; nt++)
      acc[nt] = WMMA_BF16(a1, b1[nt], acc[nt]);
  }

  // ||psi||^2 across all 256 columns (re+im): intra-half shfl reduce, cross-wave via LDS
  float s2[8];
  #pragma unroll
  for (int r = 0; r < 8; r++) {
    float v = 0;
    #pragma unroll
    for (int nt = 0; nt < 4; nt++) { float x = acc[nt][r]; v += x * x; }
    s2[r] = v;
  }
  #pragma unroll
  for (int m = 1; m < 16; m <<= 1) {
    #pragma unroll
    for (int r = 0; r < 8; r++) s2[r] += __shfl_xor(s2[r], m, 32);
  }
  __shared__ float lred[4][4][16];        // [cq][mt][row]
  if ((lane & 15) == 0) {
    #pragma unroll
    for (int r = 0; r < 8; r++) lred[cq][mt][(lane >> 4) * 8 + r] = s2[r];
  }
  __syncthreads();
  float sc[8];
  #pragma unroll
  for (int r = 0; r < 8; r++) {
    int rr = (lane >> 4) * 8 + r;
    float tot = lred[0][mt][rr] + lred[1][mt][rr] + lred[2][mt][rr] + lred[3][mt][rr];
    sc[r] = 1.0f / fmaxf(sqrtf(tot), 1e-12f);
  }
  uint16_t* ob = psib + (size_t)hh * MM * 256;
  #pragma unroll
  for (int nt = 0; nt < 4; nt++) {
    int col = cq * 64 + nt * 16 + (lane & 15);
    #pragma unroll
    for (int r = 0; r < 8; r++) {
      int row = rowbase + (lane >> 4) * 8 + r;
      ob[(size_t)row * 256 + col] = f2bf(acc[nt][r] * sc[r]);
    }
  }
}

// p = 0.95*sigmoid(w2 . silu(h . w1 + b1) + b2)  -- one wave per (row, head)
__global__ void __launch_bounds__(256, 1)
k_p(const float* __restrict__ h, const float* __restrict__ pw1,
    const float* __restrict__ pb1, const float* __restrict__ pw2,
    const float* __restrict__ pb2, float* __restrict__ pout) {
  int tid = threadIdx.x, lane = tid & 31, w = tid >> 5;
  int row = blockIdx.x * 8 + w;
  int hh = blockIdx.y;
  const float* hrow = h + (size_t)row * DD;
  const float* w1 = pw1 + (size_t)hh * DD * 16;
  float acc[16] = {0};
  for (int d = lane; d < DD; d += 32) {
    float hv = hrow[d];
    const float* wv = w1 + (size_t)d * 16;
    #pragma unroll
    for (int k = 0; k < 16; k++) acc[k] += hv * wv[k];
  }
  #pragma unroll
  for (int m = 1; m < 32; m <<= 1) {
    #pragma unroll
    for (int k = 0; k < 16; k++) acc[k] += __shfl_xor(acc[k], m, 32);
  }
  float logit = 0;
  #pragma unroll
  for (int k = 0; k < 16; k++) {
    float x = acc[k] + pb1[hh * 16 + k];
    float sg = 1.0f / (1.0f + __expf(-x));
    logit += (x * sg) * pw2[hh * 16 + k];
  }
  float pv = 0.95f / (1.0f + __expf(-(logit + pb2[hh])));
  if (lane == 0) pout[(size_t)row * HH + hh] = pv;
}

// Born-rule attention: ov = psi @ memH (2 real WMMA GEMMs, K=256), fused
// overlap^2, decoherence mix, row normalization. grid (MM/128, HH), 8 waves/block.
// Whole 16x256 A row-tile preloaded (8 frags); B streams fully unrolled.
__global__ void __launch_bounds__(256, 1)
k_attn(const uint16_t* __restrict__ psib, const uint16_t* __restrict__ mpack,
       const float* __restrict__ pbuf, uint16_t* __restrict__ attn) {
  int tid = threadIdx.x, lane = tid & 31, w = tid >> 5;
  int hh = blockIdx.y;
  int rowbase = blockIdx.x * 128 + w * 16;
  const uint16_t* pa = psib + (size_t)hh * MM * 256
                     + (size_t)(rowbase + (lane & 15)) * 256 + (lane >> 4) * 8;
  const uint16_t* mpR = mpack + (size_t)(hh * 2 + 0) * 8 * 4 * 32 * 16;
  const uint16_t* mpI = mpack + (size_t)(hh * 2 + 1) * 8 * 4 * 32 * 16;

  bf16x16 A[8];
  #pragma unroll
  for (int kt = 0; kt < 8; kt++) A[kt] = ld_frag(pa + kt * 32, pa + kt * 32 + 16);

  v8f aR[4] = {}, aI[4] = {};
  #pragma unroll
  for (int nt = 0; nt < 4; nt++) {
    #pragma unroll
    for (int kt = 0; kt < 8; kt++) {
      bf16x16 bR = ld_frag32(mpR + ((size_t)(kt * 4 + nt) * 32 + lane) * 16);
      bf16x16 bI = ld_frag32(mpI + ((size_t)(kt * 4 + nt) * 32 + lane) * 16);
      aR[nt] = WMMA_BF16(A[kt], bR, aR[nt]);
      aI[nt] = WMMA_BF16(A[kt], bI, aI[nt]);
    }
  }

  float prob[8], rs[8], raw[4][8];
  #pragma unroll
  for (int r = 0; r < 8; r++)
    prob[r] = pbuf[(size_t)(rowbase + (lane >> 4) * 8 + r) * HH + hh];
  #pragma unroll
  for (int r = 0; r < 8; r++) {
    float sum = 0;
    #pragma unroll
    for (int nt = 0; nt < 4; nt++) {
      float re = aR[nt][r], im = aI[nt][r];
      float rv = prob[r] * (re * re + im * im) + (1.0f - prob[r]) * (1.0f / 128.0f);
      raw[nt][r] = rv; sum += rv;
    }
    rs[r] = sum;
  }
  #pragma unroll
  for (int m = 1; m < 16; m <<= 1) {
    #pragma unroll
    for (int r = 0; r < 8; r++) rs[r] += __shfl_xor(rs[r], m, 32);
  }
  #pragma unroll
  for (int r = 0; r < 8; r++) {
    float inv = 1.0f / (rs[r] + 1e-8f);
    int row = rowbase + (lane >> 4) * 8 + r;
    #pragma unroll
    for (int nt = 0; nt < 4; nt++) {
      int sc = nt * 16 + (lane & 15);
      attn[(size_t)row * 512 + hh * 64 + sc] = f2bf(raw[nt][r] * inv);
    }
  }
}

// gates from attn via vg, softmax over heads, scale -> wattn (one wave per row)
__global__ void __launch_bounds__(256, 1)
k_gate(const uint16_t* __restrict__ attn, const float* __restrict__ vg,
       const float* __restrict__ gb, uint16_t* __restrict__ wattn) {
  int tid = threadIdx.x, lane = tid & 31, w = tid >> 5;
  int row = blockIdx.x * 8 + w;
  const uint16_t* arow = attn + (size_t)row * 512;
  float a[16], lg[8] = {0};
  #pragma unroll
  for (int j = 0; j < 16; j++) {
    int c = lane + 32 * j;
    a[j] = bf2f(arow[c]);
    const float* vgp = vg + (size_t)c * 8;
    #pragma unroll
    for (int g = 0; g < 8; g++) lg[g] += a[j] * vgp[g];
  }
  #pragma unroll
  for (int m = 1; m < 32; m <<= 1) {
    #pragma unroll
    for (int g = 0; g < 8; g++) lg[g] += __shfl_xor(lg[g], m, 32);
  }
  float mx = -1e30f;
  #pragma unroll
  for (int g = 0; g < 8; g++) { lg[g] += gb[g]; mx = fmaxf(mx, lg[g]); }
  float gate[8], se = 0;
  #pragma unroll
  for (int g = 0; g < 8; g++) { gate[g] = __expf(lg[g] - mx); se += gate[g]; }
  float inv = 1.0f / se;
  uint16_t* orow = wattn + (size_t)row * 512;
  #pragma unroll
  for (int j = 0; j < 16; j++) {
    int c = lane + 32 * j;
    orow[c] = f2bf(a[j] * gate[c >> 6] * inv);
  }
}

// combined = wattn(16384x512) @ Vstack(512x1024)  (K=512 WMMA GEMM) -> bf16
// block: 512 thr = 16 waves, each 16 rows x 64 cols; ping-pong pipelined.
__global__ void __launch_bounds__(512, 1)
k_comb(const uint16_t* __restrict__ wattn, const uint16_t* __restrict__ vpack,
       uint16_t* __restrict__ comb) {
  int tid = threadIdx.x, lane = tid & 31, w = tid >> 5;
  int rowbase = blockIdx.x * 16;
  const uint16_t* pa = wattn + (size_t)(rowbase + (lane & 15)) * 512 + (lane >> 4) * 8;

  v8f acc[4] = {};
  bf16x16 a0, a1, b0[4], b1[4];
  a0 = ld_frag(pa, pa + 16);
  #pragma unroll
  for (int nt = 0; nt < 4; nt++)
    b0[nt] = ld_frag32(vpack + ((size_t)(w * 4 + nt) * 32 + lane) * 16);

  for (int kt = 0; kt < 16; kt += 2) {
    int k1 = kt + 1;
    a1 = ld_frag(pa + k1 * 32, pa + k1 * 32 + 16);
    #pragma unroll
    for (int nt = 0; nt < 4; nt++)
      b1[nt] = ld_frag32(vpack + ((size_t)(k1 * 64 + w * 4 + nt) * 32 + lane) * 16);
    #pragma unroll
    for (int nt = 0; nt < 4; nt++)
      acc[nt] = WMMA_BF16(a0, b0[nt], acc[nt]);
    int k2 = (kt + 2 < 16) ? kt + 2 : 15;
    a0 = ld_frag(pa + k2 * 32, pa + k2 * 32 + 16);
    #pragma unroll
    for (int nt = 0; nt < 4; nt++)
      b0[nt] = ld_frag32(vpack + ((size_t)(k2 * 64 + w * 4 + nt) * 32 + lane) * 16);
    #pragma unroll
    for (int nt = 0; nt < 4; nt++)
      acc[nt] = WMMA_BF16(a1, b1[nt], acc[nt]);
  }

  #pragma unroll
  for (int nt = 0; nt < 4; nt++) {
    int col = w * 64 + nt * 16 + (lane & 15);
    #pragma unroll
    for (int r = 0; r < 8; r++) {
      int row = rowbase + (lane >> 4) * 8 + r;
      comb[(size_t)row * DD + col] = f2bf(acc[nt][r]);
    }
  }
}

// out = combined @ out_w + out_b  (K=1024 WMMA GEMM) -> fp32 d_out
// block: 512 thr = 16 waves, each 16 rows x 64 cols; ping-pong pipelined.
__global__ void __launch_bounds__(512, 1)
k_out(const uint16_t* __restrict__ comb, const uint16_t* __restrict__ opack,
      const float* __restrict__ ob, float* __restrict__ out) {
  int tid = threadIdx.x, lane = tid & 31, w = tid >> 5;
  int rowbase = blockIdx.x * 16;
  const uint16_t* pa = comb + (size_t)(rowbase + (lane & 15)) * DD + (lane >> 4) * 8;

  v8f acc[4] = {};
  bf16x16 a0, a1, b0[4], b1[4];
  a0 = ld_frag(pa, pa + 16);
  #pragma unroll
  for (int nt = 0; nt < 4; nt++)
    b0[nt] = ld_frag32(opack + ((size_t)(w * 4 + nt) * 32 + lane) * 16);

  for (int kt = 0; kt < 32; kt += 2) {
    int k1 = kt + 1;
    __builtin_prefetch(pa + (kt + 2) * 32, 0, 1);   // global_prefetch_b8 ahead of pipeline
    a1 = ld_frag(pa + k1 * 32, pa + k1 * 32 + 16);
    #pragma unroll
    for (int nt = 0; nt < 4; nt++)
      b1[nt] = ld_frag32(opack + ((size_t)(k1 * 64 + w * 4 + nt) * 32 + lane) * 16);
    #pragma unroll
    for (int nt = 0; nt < 4; nt++)
      acc[nt] = WMMA_BF16(a0, b0[nt], acc[nt]);
    int k2 = (kt + 2 < 32) ? kt + 2 : 31;
    a0 = ld_frag(pa + k2 * 32, pa + k2 * 32 + 16);
    #pragma unroll
    for (int nt = 0; nt < 4; nt++)
      b0[nt] = ld_frag32(opack + ((size_t)(k2 * 64 + w * 4 + nt) * 32 + lane) * 16);
    #pragma unroll
    for (int nt = 0; nt < 4; nt++)
      acc[nt] = WMMA_BF16(a1, b1[nt], acc[nt]);
  }

  #pragma unroll
  for (int nt = 0; nt < 4; nt++) {
    int col = w * 64 + nt * 16 + (lane & 15);
    float bias = ob[col];
    #pragma unroll
    for (int r = 0; r < 8; r++) {
      int row = rowbase + (lane >> 4) * 8 + r;
      out[(size_t)row * DD + col] = acc[nt][r] + bias;
    }
  }
}

// ================= host launcher =================
extern "C" void kernel_launch(void* const* d_in, const int* in_sizes, int n_in,
                              void* d_out, int out_size, void* d_ws, size_t ws_size,
                              hipStream_t stream) {
  const float* h      = (const float*)d_in[0];
  const float* W_real = (const float*)d_in[1];
  const float* W_imag = (const float*)d_in[2];
  const float* p_w1   = (const float*)d_in[3];
  const float* p_b1   = (const float*)d_in[4];
  const float* p_w2   = (const float*)d_in[5];
  const float* p_b2   = (const float*)d_in[6];
  const float* m_real = (const float*)d_in[7];
  const float* m_imag = (const float*)d_in[8];
  const float* values = (const float*)d_in[9];
  const float* gate_w = (const float*)d_in[10];
  const float* gate_b = (const float*)d_in[11];
  const float* out_w  = (const float*)d_in[12];
  const float* out_b  = (const float*)d_in[13];
  float* out = (float*)d_out;
  char* ws = (char*)d_ws;

  // workspace layout (bytes), all 32B-aligned
  size_t off_h    = 0;                                            // h bf16: 32 MB
  size_t off_wp   = off_h    + (size_t)MM * DD * 2;               // Wpack: 4 MB
  size_t off_memn = off_wp   + (size_t)HH * 32 * 16 * 32 * 16 * 2;// mem normalized f32
  size_t off_mp   = off_memn + (size_t)HH * SS * HD * 2 * 4;      // mempack bf16
  size_t off_psi  = off_mp   + (size_t)HH * 2 * 8 * 4 * 32 * 16 * 2; // psi bf16: 64 MB
  size_t off_p    = off_psi  + (size_t)HH * MM * 256 * 2;         // p f32
  size_t off_attn = off_p    + (size_t)MM * HH * 4;               // attn bf16: 16 MB
  size_t off_wat  = off_attn + (size_t)MM * 512 * 2;              // wattn bf16: 16 MB
  size_t off_vg   = off_wat  + (size_t)MM * 512 * 2;              // vg f32
  size_t off_vp   = off_vg   + (size_t)512 * 8 * 4;               // Vpack bf16
  size_t off_op   = off_vp   + (size_t)16 * 64 * 32 * 16 * 2;     // out_w pack bf16
  size_t off_comb = off_op   + (size_t)32 * 64 * 32 * 16 * 2;     // combined bf16: 32 MB

  uint16_t* hbf   = (uint16_t*)(ws + off_h);
  uint16_t* wpack = (uint16_t*)(ws + off_wp);
  float*    memn  = (float*)   (ws + off_memn);
  uint16_t* mpack = (uint16_t*)(ws + off_mp);
  uint16_t* psib  = (uint16_t*)(ws + off_psi);
  float*    pbuf  = (float*)   (ws + off_p);
  uint16_t* attn  = (uint16_t*)(ws + off_attn);
  uint16_t* watt  = (uint16_t*)(ws + off_wat);
  float*    vg    = (float*)   (ws + off_vg);
  uint16_t* vpack = (uint16_t*)(ws + off_vp);
  uint16_t* opack = (uint16_t*)(ws + off_op);
  uint16_t* comb  = (uint16_t*)(ws + off_comb);

  // ---- prep ----
  k_cvt_h    <<<MM * DD / (4 * 256), 256, 0, stream>>>(h, hbf);
  k_pack_W   <<<512, 256, 0, stream>>>(W_real, W_imag, wpack);
  k_norm_mem <<<HH * SS, 128, 0, stream>>>(m_real, m_imag, memn);
  k_pack_mem <<<64, 256, 0, stream>>>(memn, mpack);
  k_vg       <<<HH * SS, 256, 0, stream>>>(values, gate_w, vg);
  k_pack_V   <<<128, 256, 0, stream>>>(values, vpack);
  k_pack_outw<<<256, 256, 0, stream>>>(out_w, opack);

  // ---- main pipeline ----
  k_psi  <<<dim3(MM / 64, HH), 512, 0, stream>>>(hbf, wpack, psib);
  k_p    <<<dim3(MM / 8,  HH), 256, 0, stream>>>(h, p_w1, p_b1, p_w2, p_b2, pbuf);
  k_attn <<<dim3(MM / 128, HH), 256, 0, stream>>>(psib, mpack, pbuf, attn);
  k_gate <<<MM / 8, 256, 0, stream>>>(attn, vg, gate_b, watt);
  k_comb <<<MM / 16, 512, 0, stream>>>(watt, vpack, comb);
  k_out  <<<MM / 16, 512, 0, stream>>>(comb, opack, out_b, out);
}
